// DKT_11046655885804
// MI455X (gfx1250) — matrix operand order repaired and checked by reading, not compiled, not run on hardware
//
#include <hip/hip_runtime.h>
#include <hip/hip_bf16.h>

// ---------------------------------------------------------------------------
// Types for CDNA5 WMMA (wave32): 16x16x32 bf16 -> f32
// ---------------------------------------------------------------------------
typedef __attribute__((ext_vector_type(16))) __bf16 v16bf;
typedef __attribute__((ext_vector_type(8)))  __bf16 v8bf;
typedef __attribute__((ext_vector_type(8)))  float  v8f;

union Frag16 { v16bf v; v8bf h[2]; };

__device__ __forceinline__ unsigned short f2bf(float f) {
  unsigned u = __float_as_uint(f);
  u += 0x7FFFu + ((u >> 16) & 1u);          // round-to-nearest-even
  return (unsigned short)(u >> 16);
}
__device__ __forceinline__ float sigmoidf_(float x) {
  return 1.0f / (1.0f + __expf(-x));
}
__device__ __forceinline__ v8f wmma_bf16(const Frag16& a, const Frag16& b, v8f c) {
  return __builtin_amdgcn_wmma_f32_16x16x32_bf16(
      false, a.v, false, b.v, (short)0, c, false, false);
}

// ---------------------------------------------------------------------------
// Generic GEMM: C[M,N] = op( A[M,K] @ B + bias ), fp32 in/out, bf16 WMMA core.
//   transB==0 : B is [K,N] row-major (ldb)
//   transB==1 : B is [N,K] row-major (ldb)   (i.e. computes A @ B^T)
//   relu      : apply max(0,.)
//   accum     : C += result (reads existing C)
// Tile: 128x128x32, 256 threads = 8 waves, wave (wm in 0..1, wn in 0..3)
// owns a 64x32 region = 4x2 WMMA tiles.
// ---------------------------------------------------------------------------
#define BM 128
#define BN 128
#define BK 32
#define LDSS 40   // padded bf16 row stride (bank-conflict-free b128 frag loads)

__global__ __launch_bounds__(256)
void gemm_bf16_wmma(const float* __restrict__ A, int lda,
                    const float* __restrict__ B, int ldb, int transB,
                    const float* __restrict__ bias,
                    float* __restrict__ C, int ldc,
                    int M, int N, int K, int relu, int accum)
{
  __shared__ unsigned short As[BM * LDSS];
  __shared__ unsigned short Bs[BN * LDSS];

  const int tid   = threadIdx.x;
  const int lane  = tid & 31;
  const int w     = tid >> 5;
  const int waveM = w >> 2;          // 0..1
  const int waveN = w & 3;           // 0..3
  const int lrow  = lane & 15;
  const int lhalf = lane >> 4;

  const int rowBase = blockIdx.y * BM;
  const int colBase = blockIdx.x * BN;

  const int r0 = tid >> 3;           // 0..31
  const int cg = (tid & 7) * 4;      // 0..28 (k column group of 4)

  v8f acc[4][2];
#pragma unroll
  for (int i = 0; i < 4; ++i)
#pragma unroll
    for (int j = 0; j < 2; ++j)
#pragma unroll
      for (int e = 0; e < 8; ++e) acc[i][j][e] = 0.0f;

  for (int kb = 0; kb < K; kb += BK) {
    // ---- stage A tile (128 x 32) fp32 -> bf16 LDS ----
#pragma unroll
    for (int p = 0; p < 4; ++p) {
      int r  = p * 32 + r0;
      int gr = rowBase + r;
      int gk = kb + cg;
      float x0 = 0.f, x1 = 0.f, x2 = 0.f, x3 = 0.f;
      if (gr < M) {
        if (gk + 4 <= K) {
          const float4 t = *(const float4*)(A + (size_t)gr * lda + gk);
          x0 = t.x; x1 = t.y; x2 = t.z; x3 = t.w;
        } else {
          if (gk + 0 < K) x0 = A[(size_t)gr * lda + gk + 0];
          if (gk + 1 < K) x1 = A[(size_t)gr * lda + gk + 1];
          if (gk + 2 < K) x2 = A[(size_t)gr * lda + gk + 2];
          if (gk + 3 < K) x3 = A[(size_t)gr * lda + gk + 3];
        }
      }
      ushort4 s; s.x = f2bf(x0); s.y = f2bf(x1); s.z = f2bf(x2); s.w = f2bf(x3);
      *(ushort4*)&As[r * LDSS + cg] = s;
    }
    // ---- stage B tile into Bs[n][k] (bf16) ----
    if (transB) {
#pragma unroll
      for (int p = 0; p < 4; ++p) {
        int n  = p * 32 + r0;
        int gn = colBase + n;
        int gk = kb + cg;
        float x0 = 0.f, x1 = 0.f, x2 = 0.f, x3 = 0.f;
        if (gn < N) {
          if (gk + 4 <= K) {
            const float4 t = *(const float4*)(B + (size_t)gn * ldb + gk);
            x0 = t.x; x1 = t.y; x2 = t.z; x3 = t.w;
          } else {
            if (gk + 0 < K) x0 = B[(size_t)gn * ldb + gk + 0];
            if (gk + 1 < K) x1 = B[(size_t)gn * ldb + gk + 1];
            if (gk + 2 < K) x2 = B[(size_t)gn * ldb + gk + 2];
            if (gk + 3 < K) x3 = B[(size_t)gn * ldb + gk + 3];
          }
        }
        ushort4 s; s.x = f2bf(x0); s.y = f2bf(x1); s.z = f2bf(x2); s.w = f2bf(x3);
        *(ushort4*)&Bs[n * LDSS + cg] = s;
      }
    } else {
      int n  = tid & 127;
      int kh = (tid >> 7) * 16;
      int gn = colBase + n;
#pragma unroll
      for (int kk = 0; kk < 16; ++kk) {
        int k  = kh + kk;
        int gk = kb + k;
        float x = 0.f;
        if (gn < N && gk < K) x = B[(size_t)gk * ldb + gn];
        Bs[n * LDSS + k] = f2bf(x);
      }
    }
    __syncthreads();

    // ---- fragments + WMMA ----
    Frag16 af[4], bf[2];
#pragma unroll
    for (int mt = 0; mt < 4; ++mt) {
      int r = waveM * 64 + mt * 16 + lrow;
      af[mt].h[0] = *(const v8bf*)&As[r * LDSS + lhalf * 8];        // k = 8h+0..7
      af[mt].h[1] = *(const v8bf*)&As[r * LDSS + 16 + lhalf * 8];   // k = 16+8h+0..7
    }
#pragma unroll
    for (int nt = 0; nt < 2; ++nt) {
      int n = waveN * 32 + nt * 16 + lrow;
      bf[nt].h[0] = *(const v8bf*)&Bs[n * LDSS + lhalf * 8];        // k = 8h+0..7
      bf[nt].h[1] = *(const v8bf*)&Bs[n * LDSS + 16 + lhalf * 8];   // k = 16+8h+0..7
    }
#pragma unroll
    for (int mt = 0; mt < 4; ++mt)
#pragma unroll
      for (int nt = 0; nt < 2; ++nt)
        acc[mt][nt] = wmma_bf16(af[mt], bf[nt], acc[mt][nt]);
    __syncthreads();
  }

  // ---- epilogue ----
#pragma unroll
  for (int mt = 0; mt < 4; ++mt) {
#pragma unroll
    for (int nt = 0; nt < 2; ++nt) {
      int col = colBase + waveN * 32 + nt * 16 + lrow;
      float bv = 0.0f;
      if (bias != nullptr && col < N) bv = bias[col];
#pragma unroll
      for (int v = 0; v < 8; ++v) {
        int row = rowBase + waveM * 64 + mt * 16 + v + 8 * lhalf;
        if (row < M && col < N) {
          float val = acc[mt][nt][v] + bv;
          if (accum) val += C[(size_t)row * ldc + col];
          if (relu)  val = fmaxf(val, 0.0f);
          C[(size_t)row * ldc + col] = val;
        }
      }
    }
  }
}

// ---------------------------------------------------------------------------
// GRU recurrence (torch gate order r,z,n), one workgroup per GRU instance.
// xp: [B*L, 3H] precomputed x@Wih^T + bih. Whh pre-converted to bf16 [3H,H].
// Wave w owns hidden columns [16w,16w+16) and its r/z/n gate tiles, so the
// n-gate's dependence on r is lane-local in the WMMA D-layout.
// Per step per wave: 4 M-tiles x 3 gates x 4 k-steps = 48 bf16 WMMAs.
// ---------------------------------------------------------------------------
#define HST 136   // padded bf16 row stride for h in LDS

__global__ __launch_bounds__(256)
void gru_wmma(const float* __restrict__ xp1, const unsigned short* __restrict__ W1,
              const float* __restrict__ bhh1, float* __restrict__ out1,
              const float* __restrict__ xp2, const unsigned short* __restrict__ W2,
              const float* __restrict__ bhh2, float* __restrict__ out2)
{
  const float*          xp  = (blockIdx.x == 0) ? xp1  : xp2;
  const unsigned short* W   = (blockIdx.x == 0) ? W1   : W2;
  const float*          bhh = (blockIdx.x == 0) ? bhh1 : bhh2;
  float*                out = (blockIdx.x == 0) ? out1 : out2;

  __shared__ unsigned short hbf[64 * HST];

  const int tid   = threadIdx.x;
  const int lane  = tid & 31;
  const int w     = tid >> 5;        // 0..7 -> hidden column group
  const int lrow  = lane & 15;
  const int lhalf = lane >> 4;
  const int c     = w * 16 + lrow;   // hidden column 0..127

  for (int i = tid; i < 64 * HST; i += 256) hbf[i] = 0;

  // Persistent Whh fragments: [gate][kstep], held in registers for all steps.
  Frag16 bfrag[3][4];
#pragma unroll
  for (int g = 0; g < 3; ++g) {
#pragma unroll
    for (int kb = 0; kb < 4; ++kb) {
      int n    = g * 128 + c;
      int base = n * 128 + kb * 32 + lhalf * 8;
      bfrag[g][kb].h[0] = *(const v8bf*)&W[base];
      bfrag[g][kb].h[1] = *(const v8bf*)&W[base + 16];
    }
  }
  const float br = bhh[c], bz = bhh[128 + c], bn = bhh[256 + c];

  float hreg[4][8];
#pragma unroll
  for (int mt = 0; mt < 4; ++mt)
#pragma unroll
    for (int v = 0; v < 8; ++v) hreg[mt][v] = 0.0f;

  __syncthreads();

  for (int t = 0; t < 200; ++t) {
    v8f racc[4], zacc[4], nacc[4];
#pragma unroll
    for (int mt = 0; mt < 4; ++mt)
#pragma unroll
      for (int e = 0; e < 8; ++e) { racc[mt][e] = 0.f; zacc[mt][e] = 0.f; nacc[mt][e] = 0.f; }

#pragma unroll
    for (int mt = 0; mt < 4; ++mt) {
      int r = mt * 16 + lrow;
#pragma unroll
      for (int kb = 0; kb < 4; ++kb) {
        Frag16 af;
        af.h[0] = *(const v8bf*)&hbf[r * HST + kb * 32 + lhalf * 8];
        af.h[1] = *(const v8bf*)&hbf[r * HST + kb * 32 + 16 + lhalf * 8];
        racc[mt] = wmma_bf16(af, bfrag[0][kb], racc[mt]);
        zacc[mt] = wmma_bf16(af, bfrag[1][kb], zacc[mt]);
        nacc[mt] = wmma_bf16(af, bfrag[2][kb], nacc[mt]);
      }
    }
    __syncthreads();   // all hbf reads done before overwrite

#pragma unroll
    for (int mt = 0; mt < 4; ++mt) {
#pragma unroll
      for (int v = 0; v < 8; ++v) {
        int b = mt * 16 + v + 8 * lhalf;                 // batch row 0..63
        size_t xi = ((size_t)b * 200 + t) * 384;
        float rg = sigmoidf_(xp[xi + c]       + racc[mt][v] + br);
        float zg = sigmoidf_(xp[xi + 128 + c] + zacc[mt][v] + bz);
        float ng = tanhf   (xp[xi + 256 + c] + rg * (nacc[mt][v] + bn));
        float h  = (1.0f - zg) * ng + zg * hreg[mt][v];
        hreg[mt][v] = h;
        out[(size_t)b * 25600 + (size_t)t * 128 + c] = h;
        hbf[b * HST + c] = f2bf(h);
      }
    }
    __syncthreads();   // hbf updates visible before next step's reads
  }
}

// ---------------------------------------------------------------------------
// Small elementwise kernels
// ---------------------------------------------------------------------------
__global__ void convert_whh(const float* __restrict__ w1, const float* __restrict__ w2,
                            unsigned short* __restrict__ o1, unsigned short* __restrict__ o2,
                            int n) {
  int i = blockIdx.x * 256 + threadIdx.x;
  if (i < n) o1[i] = f2bf(w1[i]);
  else { int j = i - n; if (j < n) o2[j] = f2bf(w2[j]); }
}

__global__ void gate_cat(const float* __restrict__ ts, const float* __restrict__ oh,
                         const float* __restrict__ od, float* __restrict__ cat, int total) {
  int i = blockIdx.x * 256 + threadIdx.x;
  if (i >= total) return;
  int row = i >> 7, c = i & 127;
  float th = sigmoidf_(ts[i]);
  cat[(size_t)row * 256 + c]       = th * od[i];
  cat[(size_t)row * 256 + 128 + c] = (1.0f - th) * oh[i];
}

// ---------------------------------------------------------------------------
// Host orchestration
// ---------------------------------------------------------------------------
extern "C" void kernel_launch(void* const* d_in, const int* in_sizes, int n_in,
                              void* d_out, int out_size, void* d_ws, size_t ws_size,
                              hipStream_t stream)
{
  (void)in_sizes; (void)n_in; (void)out_size; (void)ws_size;
  const float* x       = (const float*)d_in[0];
  const float* ques    = (const float*)d_in[1];
  const float* G       = (const float*)d_in[2];
  const float* adj_in  = (const float*)d_in[3];
  const float* adj_out = (const float*)d_in[4];
  const float* hg_W1 = (const float*)d_in[5];  const float* hg_b1 = (const float*)d_in[6];
  const float* hg_W2 = (const float*)d_in[7];  const float* hg_b2 = (const float*)d_in[8];
  const float* g1_W1 = (const float*)d_in[9];  const float* g1_b1 = (const float*)d_in[10];
  const float* g1_W2 = (const float*)d_in[11]; const float* g1_b2 = (const float*)d_in[12];
  const float* g2_W1 = (const float*)d_in[13]; const float* g2_b1 = (const float*)d_in[14];
  const float* g2_W2 = (const float*)d_in[15]; const float* g2_b2 = (const float*)d_in[16];
  const float* r1_Wih = (const float*)d_in[17]; const float* r1_Whh = (const float*)d_in[18];
  const float* r1_bih = (const float*)d_in[19]; const float* r1_bhh = (const float*)d_in[20];
  const float* r2_Wih = (const float*)d_in[21]; const float* r2_Whh = (const float*)d_in[22];
  const float* r2_bih = (const float*)d_in[23]; const float* r2_bhh = (const float*)d_in[24];
  const float* fcc_W = (const float*)d_in[25]; const float* fcc_b = (const float*)d_in[26];
  const float* fct_W = (const float*)d_in[27]; const float* fct_b = (const float*)d_in[28];
  const float* fce_W = (const float*)d_in[29]; const float* fce_b = (const float*)d_in[30];
  const float* w1_W  = (const float*)d_in[31]; const float* w1_b  = (const float*)d_in[32];
  const float* w2_W  = (const float*)d_in[33]; const float* w2_b  = (const float*)d_in[34];

  float* ws  = (float*)d_ws;
  float* t1  = ws;                    // 5000x128
  float* t2  = ws + 640000;           // 5000x128
  float* qh  = ws + 1280000;          // 5000x128
  float* qd  = ws + 1920000;          // 5000x128 (concat target)
  float* xh  = ws + 2560000;          // 12800x128
  float* xd  = ws + 4198400;          // 12800x128
  float* xp1 = ws + 5836800;          // 12800x384
  float* xp2 = ws + 10752000;         // 12800x384
  float* oh  = ws + 15667200;         // 12800x128
  float* od  = ws + 17305600;         // 12800x128
  float* tsb = ws + 18944000;         // 12800x128 (theta pre-activation)
  float* cat = ws + 20582400;         // 12800x256
  unsigned short* w1bf = (unsigned short*)(ws + 23859200);  // 384x128 bf16
  unsigned short* w2bf = w1bf + 49152;

  float* out = (float*)d_out;         // [logit_c | logit_t | ensemble], 32M each

  auto gemm = [&](const float* A, int lda, const float* Bm, int ldb, int tB,
                  const float* bias, float* C, int ldc,
                  int M, int N, int K, int relu, int accum) {
    dim3 grid((unsigned)((N + 127) / 128), (unsigned)((M + 127) / 128));
    gemm_bf16_wmma<<<grid, 256, 0, stream>>>(A, lda, Bm, ldb, tB, bias, C, ldc,
                                             M, N, K, relu, accum);
  };

  convert_whh<<<(2 * 49152 + 255) / 256, 256, 0, stream>>>(r1_Whh, r2_Whh, w1bf, w2bf, 49152);

  // ---- SkillGraph HGCN: ques_h = G @ (relu(G @ (ques@W1+b1)) @ W2 + b2) ----
  gemm(ques, 128, hg_W1, 128, 0, hg_b1, t1, 128, 5000, 128, 128, 0, 0);
  gemm(G, 5000, t1, 128, 0, nullptr, t2, 128, 5000, 128, 5000, 1, 0);
  gemm(t2, 128, hg_W2, 128, 0, hg_b2, t1, 128, 5000, 128, 128, 0, 0);
  gemm(G, 5000, t1, 128, 0, nullptr, qh, 128, 5000, 128, 5000, 0, 0);
  // ---- TransitionGraph g1 (adj_out) -> ques_out -> qd[:,64:128] ----
  gemm(ques, 128, g1_W1, 128, 0, g1_b1, t1, 128, 5000, 128, 128, 0, 0);
  gemm(adj_out, 5000, t1, 128, 0, nullptr, t2, 128, 5000, 128, 5000, 1, 0);
  gemm(t2, 128, g1_W2, 64, 0, g1_b2, t1, 64, 5000, 64, 128, 0, 0);
  gemm(adj_out, 5000, t1, 64, 0, nullptr, qd + 64, 128, 5000, 64, 5000, 0, 0);
  // ---- TransitionGraph g2 (adj_in) -> ques_in -> qd[:,0:64] ----
  gemm(ques, 128, g2_W1, 128, 0, g2_b1, t1, 128, 5000, 128, 128, 0, 0);
  gemm(adj_in, 5000, t1, 128, 0, nullptr, t2, 128, 5000, 128, 5000, 1, 0);
  gemm(t2, 128, g2_W2, 64, 0, g2_b2, t1, 64, 5000, 64, 128, 0, 0);
  gemm(adj_in, 5000, t1, 64, 0, nullptr, qd, 128, 5000, 64, 5000, 0, 0);
  // ---- soft selection ----
  gemm(x, 5000, qh, 128, 0, nullptr, xh, 128, 12800, 128, 5000, 0, 0);
  gemm(x, 5000, qd, 128, 0, nullptr, xd, 128, 12800, 128, 5000, 0, 0);
  // ---- GRU input projections (A @ Wih^T + bih) ----
  gemm(xh, 128, r1_Wih, 128, 1, r1_bih, xp1, 384, 12800, 384, 128, 0, 0);
  gemm(xd, 128, r2_Wih, 128, 1, r2_bih, xp2, 384, 12800, 384, 128, 0, 0);
  // ---- dual GRU recurrence (2 persistent workgroups) ----
  gru_wmma<<<2, 256, 0, stream>>>(xp1, w1bf, r1_bhh, oh, xp2, w2bf, r2_bhh, od);
  // ---- heads ----
  gemm(oh, 128, fcc_W, 128, 1, fcc_b, out,            2500, 12800, 2500, 128, 0, 0);
  gemm(od, 128, fct_W, 128, 1, fct_b, out + 32000000, 2500, 12800, 2500, 128, 0, 0);
  gemm(oh, 128, w1_W, 128, 1, w1_b, tsb, 128, 12800, 128, 128, 0, 0);
  gemm(od, 128, w2_W, 128, 1, w2_b, tsb, 128, 12800, 128, 128, 0, 1);   // accumulate
  gate_cat<<<(12800 * 128 + 255) / 256, 256, 0, stream>>>(tsb, oh, od, cat, 12800 * 128);
  gemm(cat, 256, fce_W, 256, 1, fce_b, out + 64000000, 2500, 12800, 2500, 256, 0, 0);
}
